// MultiHeadAttention_32727650796273
// MI455X (gfx1250) — compile-verified, matrix operand order checked by reading, not run on hardware
//
#include <hip/hip_runtime.h>
#include <hip/hip_bf16.h>
#include <math.h>

// ---- problem constants -----------------------------------------------------
#define NH     16
#define DKV    64
#define DM     1024
#define LSEQ   2048
#define BATCH  2
#define MTOT   (BATCH * LSEQ)     // 4096 activation rows
#define EPS    1e-6f

typedef __attribute__((ext_vector_type(16))) __bf16 v16bf;
typedef __attribute__((ext_vector_type(8)))  float  v8f;
typedef int v4i32 __attribute__((vector_size(16)));   // matches builtin proto

// ---- optional CDNA5 async global->LDS path (guarded; falls back cleanly) ---
#if defined(__has_builtin)
#if __has_builtin(__builtin_amdgcn_global_load_async_to_lds_b128) && \
    __has_builtin(__builtin_amdgcn_s_wait_asynccnt)
#define USE_ASYNC_LDS 1
#endif
#endif
#ifndef USE_ASYNC_LDS
#define USE_ASYNC_LDS 0
#endif

__device__ __forceinline__ void copy16_g2l(const __bf16* __restrict__ src,
                                           __bf16* dst) {
#if USE_ASYNC_LDS
  // global_load_async_to_lds_b128: (global int4* src, shared int4* dst,
  //                                 imm offset, imm cpol)
  __builtin_amdgcn_global_load_async_to_lds_b128(
      (__attribute__((address_space(1))) v4i32*)(void*)src,
      (__attribute__((address_space(3))) v4i32*)(void*)dst, 0, 0);
#else
  *(uint4*)dst = *(const uint4*)src;
#endif
}

__device__ __forceinline__ void stage_join() {
#if USE_ASYNC_LDS
  __builtin_amdgcn_s_wait_asynccnt(0);
#endif
  __syncthreads();
}

// ---------------------------------------------------------------------------
// WMMA fragment helpers (CDNA5 ISA 7.12.2 lane layouts, wave32).
// A (16x32 bf16): lane row = lane&15, kbase = (lane>>4)*8; elements 0..7 hold
// K=kbase..kbase+7, elements 8..15 hold K=kbase+16..kbase+23.  B mirrors A
// with N in place of M, so a row-major [N][K] buffer loads identically.
// ---------------------------------------------------------------------------
__device__ __forceinline__ v16bf frag_bf16(const __bf16* __restrict__ base,
                                           int ld, int row0, int col0, int lane) {
  const __bf16* p = base + (size_t)(row0 + (lane & 15)) * ld + col0 + ((lane >> 4) << 3);
  v16bf f;
#pragma unroll
  for (int i = 0; i < 8; ++i) { f[i] = p[i]; f[i + 8] = p[i + 16]; }
  return f;
}

__device__ __forceinline__ v16bf frag_f32(const float* __restrict__ base,
                                          int ld, int row0, int col0, int lane) {
  const float* p = base + (size_t)(row0 + (lane & 15)) * ld + col0 + ((lane >> 4) << 3);
  v16bf f;
#pragma unroll
  for (int i = 0; i < 8; ++i) { f[i] = (__bf16)p[i]; f[i + 8] = (__bf16)p[i + 16]; }
  return f;
}

__device__ __forceinline__ v8f wmma_bf16(v16bf a, v16bf b, v8f c) {
  return __builtin_amdgcn_wmma_f32_16x16x32_bf16(false, a, false, b, (short)0, c,
                                                 false, false);
}

// ---------------------------------------------------------------------------
// Shared GEMM core.  Block = 128 threads = 4 waves arranged 2(M) x 2(N).
// Block tile: 128(M) x 64(N).  Per wave: 64x32 = 4x2 WMMA fragments.
// B^T tile (64 rows x 32 K, bf16, 4 KB) is staged in LDS each K-step and
// shared by all waves (async global->LDS when available).
//   Asrc : A base (row 0 of the whole A matrix; m0 holds the block offset), lda in elements
//   Bsrc : B^T base pre-offset to this block's first N row, ldb in elements
//   m0   : this wave's first A row relative to Asrc
// ---------------------------------------------------------------------------
template <bool AF32>
__device__ __forceinline__ void gemm_core(const void* __restrict__ Asrc, int lda,
                                          const __bf16* __restrict__ Bsrc, int ldb,
                                          __bf16* Bs, int m0, int K,
                                          v8f acc[4][2]) {
  const int t    = threadIdx.x;
  const int lane = t & 31;
  const int nw   = ((t >> 5) & 1) << 5;     // wave's N offset inside LDS tile

  for (int k0 = 0; k0 < K; k0 += 32) {
    __syncthreads();                         // previous LDS tile fully consumed
    const int c0 = t << 1;                   // 256 x 16B chunks, 2 per thread
#pragma unroll
    for (int u = 0; u < 2; ++u) {
      const int c   = c0 + u;
      const int row = c >> 2;
      const int ch  = (c & 3) << 3;          // element offset within 32-K row
      copy16_g2l(Bsrc + (size_t)row * ldb + k0 + ch, Bs + row * 32 + ch);
    }
    stage_join();

    v16bf a[4], b[2];
#pragma unroll
    for (int i = 0; i < 4; ++i) {
      if constexpr (AF32)
        a[i] = frag_f32((const float*)Asrc, lda, m0 + i * 16, k0, lane);
      else
        a[i] = frag_bf16((const __bf16*)Asrc, lda, m0 + i * 16, k0, lane);
    }
#pragma unroll
    for (int j = 0; j < 2; ++j) b[j] = frag_bf16(Bs, 32, nw + j * 16, 0, lane);
#pragma unroll
    for (int i = 0; i < 4; ++i)
#pragma unroll
      for (int j = 0; j < 2; ++j) acc[i][j] = wmma_bf16(a[i], b[j], acc[i][j]);
  }
}

// ---------------------------------------------------------------------------
// Kernel 0: LDS-tiled transpose+convert f32 weight [K=DM][N=DM] -> bf16 [N][K]
// grid (DM/32, DM/32), block 256 (32x8)
// ---------------------------------------------------------------------------
__global__ void prep_weights(const float* __restrict__ w, __bf16* __restrict__ wt) {
  __shared__ float tile[32][33];
  const int bx = blockIdx.x * 32;            // N origin
  const int by = blockIdx.y * 32;            // K origin
  const int tx = threadIdx.x & 31;
  const int ty = threadIdx.x >> 5;
#pragma unroll
  for (int r = 0; r < 32; r += 8)
    tile[ty + r][tx] = w[(size_t)(by + ty + r) * DM + bx + tx];
  __syncthreads();
#pragma unroll
  for (int r = 0; r < 32; r += 8)
    wt[(size_t)(bx + ty + r) * DM + by + tx] = (__bf16)tile[tx][ty + r];
}

// ---------------------------------------------------------------------------
// Kernel 1: projection GEMM  Y[4096,1024] = X(f32) @ W (bf16, pre-transposed)
// VMODE==0 -> bf16 row-major [M][1024];  VMODE==1 -> per-head V^T [B][H][dv][L]
// grid (MTOT/128, DM/64), block 128
// ---------------------------------------------------------------------------
template <int VMODE>
__global__ void proj_gemm(const float* __restrict__ X, const __bf16* __restrict__ Wt,
                          __bf16* __restrict__ out) {
  __shared__ __bf16 Bs[64 * 32];
  const int lane = threadIdx.x & 31;
  const int wv   = threadIdx.x >> 5;
  const int m0   = blockIdx.x * 128 + (wv >> 1) * 64;
  const int nblk = blockIdx.y * 64;
  const int nw   = (wv & 1) << 5;

  v8f acc[4][2] = {};
  gemm_core<true>(X, DM, Wt + (size_t)nblk * DM, DM, Bs, m0, DM, acc);

  const int nn = lane & 15;
  const int mb = (lane >> 4) << 3;
#pragma unroll
  for (int i = 0; i < 4; ++i)
#pragma unroll
    for (int j = 0; j < 2; ++j) {
      const int ncol = nblk + nw + j * 16 + nn;
#pragma unroll
      for (int r = 0; r < 8; ++r) {
        const int m = m0 + i * 16 + mb + r;
        const __bf16 val = (__bf16)acc[i][j][r];
        if (VMODE == 0) {
          out[(size_t)m * DM + ncol] = val;
        } else {                              // V transposed: [B][H][dv][L]
          const int b_ = m >> 11, l = m & 2047;
          const int h  = ncol >> 6, d = ncol & 63;
          out[(((size_t)(b_ * NH + h) * DKV + d) << 11) + l] = val;
        }
      }
    }
}

// ---------------------------------------------------------------------------
// Kernel 2: S[b,h,q,k] = (Qh . Kh) / sqrt(dk), f32 into attn output region.
// grid (LSEQ/128, LSEQ/64, B*NH), block 128
// ---------------------------------------------------------------------------
__global__ void attn_scores(const __bf16* __restrict__ Qh, const __bf16* __restrict__ Kh,
                            float* __restrict__ attn) {
  __shared__ __bf16 Bs[64 * 32];
  const int lane = threadIdx.x & 31;
  const int wv   = threadIdx.x >> 5;
  const int bh   = blockIdx.z;
  const int b    = bh >> 4, h = bh & 15;
  const int m0   = blockIdx.x * 128 + (wv >> 1) * 64;   // query row (local)
  const int nblk = blockIdx.y * 64;                     // key row origin
  const int nw   = (wv & 1) << 5;

  const __bf16* A = Qh + (size_t)(b * LSEQ) * DM + h * DKV;
  const __bf16* B = Kh + (size_t)(b * LSEQ + nblk) * DM + h * DKV;

  v8f acc[4][2] = {};
  gemm_core<false>(A, DM, B, DM, Bs, m0, DKV, acc);

  const float scale = 0.125f;                            // 1/sqrt(64)
  const size_t base = (size_t)bh * LSEQ * LSEQ;
  const int nn = lane & 15;
  const int mb = (lane >> 4) << 3;
#pragma unroll
  for (int i = 0; i < 4; ++i)
#pragma unroll
    for (int j = 0; j < 2; ++j) {
      const int ncol = nblk + nw + j * 16 + nn;
#pragma unroll
      for (int r = 0; r < 8; ++r) {
        const int qq = m0 + i * 16 + mb + r;
        attn[base + (size_t)qq * LSEQ + ncol] = acc[i][j][r] * scale;
      }
    }
}

// ---------------------------------------------------------------------------
// Kernel 3: in-place row softmax over 2048 f32. One 256-thread block per row.
// ---------------------------------------------------------------------------
__global__ void softmax_rows(float* __restrict__ attn) {
  __shared__ float red[256];
  float* p = attn + (size_t)blockIdx.x * LSEQ;
  const int t = threadIdx.x;

  float vals[8];
  float mx = -INFINITY;
#pragma unroll
  for (int i = 0; i < 8; ++i) { vals[i] = p[t + 256 * i]; mx = fmaxf(mx, vals[i]); }
  red[t] = mx; __syncthreads();
  for (int s = 128; s > 0; s >>= 1) { if (t < s) red[t] = fmaxf(red[t], red[t + s]); __syncthreads(); }
  mx = red[0]; __syncthreads();

  float sum = 0.f;
#pragma unroll
  for (int i = 0; i < 8; ++i) { vals[i] = __expf(vals[i] - mx); sum += vals[i]; }
  red[t] = sum; __syncthreads();
  for (int s = 128; s > 0; s >>= 1) { if (t < s) red[t] += red[t + s]; __syncthreads(); }
  const float inv = 1.0f / red[0];
#pragma unroll
  for (int i = 0; i < 8; ++i) p[t + 256 * i] = vals[i] * inv;
}

// ---------------------------------------------------------------------------
// Kernel 4: O = attn @ V per head.  A = attn f32 (cvt on the fly),
// B = V^T bf16 [dv][L].  Output bf16 [4096][H*dv].
// grid (LSEQ/128, 1, B*NH), block 128
// ---------------------------------------------------------------------------
__global__ void attn_v(const float* __restrict__ attn, const __bf16* __restrict__ Vt,
                       __bf16* __restrict__ O) {
  __shared__ __bf16 Bs[64 * 32];
  const int lane = threadIdx.x & 31;
  const int wv   = threadIdx.x >> 5;
  const int bh   = blockIdx.z;
  const int b    = bh >> 4, h = bh & 15;
  const int m0   = blockIdx.x * 128 + (wv >> 1) * 64;
  const int nw   = (wv & 1) << 5;

  const float*  A = attn + (size_t)bh * LSEQ * LSEQ;
  const __bf16* B = Vt + ((size_t)bh * DKV << 11);       // [dv][LSEQ]

  v8f acc[4][2] = {};
  gemm_core<true>(A, LSEQ, B, LSEQ, Bs, m0, LSEQ, acc);

  const int nn = lane & 15;
  const int mb = (lane >> 4) << 3;
#pragma unroll
  for (int i = 0; i < 4; ++i)
#pragma unroll
    for (int j = 0; j < 2; ++j) {
      const int d = nw + j * 16 + nn;
#pragma unroll
      for (int r = 0; r < 8; ++r) {
        const int m = b * LSEQ + m0 + i * 16 + mb + r;
        O[(size_t)m * DM + h * DKV + d] = (__bf16)acc[i][j][r];
      }
    }
}

// ---------------------------------------------------------------------------
// Kernel 5: Y = O @ w_fc + residual(v), f32 into scratch.
// grid (MTOT/128, DM/64), block 128
// ---------------------------------------------------------------------------
__global__ void fc_gemm(const __bf16* __restrict__ O, const __bf16* __restrict__ Wt,
                        const float* __restrict__ resid, float* __restrict__ Y) {
  __shared__ __bf16 Bs[64 * 32];
  const int lane = threadIdx.x & 31;
  const int wv   = threadIdx.x >> 5;
  const int m0   = blockIdx.x * 128 + (wv >> 1) * 64;
  const int nblk = blockIdx.y * 64;
  const int nw   = (wv & 1) << 5;

  v8f acc[4][2] = {};
  gemm_core<false>(O, DM, Wt + (size_t)nblk * DM, DM, Bs, m0, DM, acc);

  const int nn = lane & 15;
  const int mb = (lane >> 4) << 3;
#pragma unroll
  for (int i = 0; i < 4; ++i)
#pragma unroll
    for (int j = 0; j < 2; ++j) {
      const int n = nblk + nw + j * 16 + nn;
#pragma unroll
      for (int r = 0; r < 8; ++r) {
        const int m = m0 + i * 16 + mb + r;
        Y[(size_t)m * DM + n] = acc[i][j][r] + resid[(size_t)m * DM + n];
      }
    }
}

// ---------------------------------------------------------------------------
// Kernel 6: row LayerNorm over 1024, one 256-thread block per row -> d_out
// ---------------------------------------------------------------------------
__global__ void layernorm_rows(const float* __restrict__ Y, const float* __restrict__ g,
                               const float* __restrict__ bta, float* __restrict__ out) {
  __shared__ float s1[256], s2[256];
  const float* p = Y + (size_t)blockIdx.x * DM;
  float* o = out + (size_t)blockIdx.x * DM;
  const int t = threadIdx.x;

  float x[4], a = 0.f, b = 0.f;
#pragma unroll
  for (int i = 0; i < 4; ++i) { x[i] = p[t + 256 * i]; a += x[i]; b += x[i] * x[i]; }
  s1[t] = a; s2[t] = b; __syncthreads();
  for (int s = 128; s > 0; s >>= 1) {
    if (t < s) { s1[t] += s1[t + s]; s2[t] += s2[t + s]; }
    __syncthreads();
  }
  const float mean = s1[0] * (1.0f / DM);
  const float var  = s2[0] * (1.0f / DM) - mean * mean;
  const float rstd = rsqrtf(var + EPS);
#pragma unroll
  for (int i = 0; i < 4; ++i) {
    const int n = t + 256 * i;
    o[n] = (x[i] - mean) * rstd * g[n] + bta[n];
  }
}

// ---------------------------------------------------------------------------
extern "C" void kernel_launch(void* const* d_in, const int* in_sizes, int n_in,
                              void* d_out, int out_size, void* d_ws, size_t ws_size,
                              hipStream_t stream) {
  const float* q    = (const float*)d_in[0];
  const float* k    = (const float*)d_in[1];
  const float* v    = (const float*)d_in[2];
  const float* w_q  = (const float*)d_in[3];
  const float* w_k  = (const float*)d_in[4];
  const float* w_v  = (const float*)d_in[5];
  const float* w_fc = (const float*)d_in[6];
  const float* ln_g = (const float*)d_in[7];
  const float* ln_b = (const float*)d_in[8];

  float* out  = (float*)d_out;                                   // [2,2048,1024]
  float* attn = out + (size_t)BATCH * LSEQ * DM;                 // [2,16,2048,2048]

  // workspace carve-up (~56 MB)
  char* ws = (char*)d_ws;
  const size_t WBYTES = (size_t)DM * DM * sizeof(__bf16);        // 2 MB
  const size_t ABYTES = (size_t)MTOT * DM * sizeof(__bf16);      // 8 MB
  __bf16* wt_q  = (__bf16*)ws;                 ws += WBYTES;
  __bf16* wt_k  = (__bf16*)ws;                 ws += WBYTES;
  __bf16* wt_v  = (__bf16*)ws;                 ws += WBYTES;
  __bf16* wt_fc = (__bf16*)ws;                 ws += WBYTES;
  __bf16* Qh    = (__bf16*)ws;                 ws += ABYTES;
  __bf16* Kh    = (__bf16*)ws;                 ws += ABYTES;
  __bf16* Vt    = (__bf16*)ws;                 ws += ABYTES;     // [B][H][dv][L]
  __bf16* O     = (__bf16*)ws;                 ws += ABYTES;
  float*  Y     = (float*)ws;                                    // 16 MB

  const dim3 blk(128);
  // 0) weight transpose + bf16 convert (coalesced LDS tile transpose)
  prep_weights<<<dim3(DM / 32, DM / 32), dim3(256), 0, stream>>>(w_q,  wt_q);
  prep_weights<<<dim3(DM / 32, DM / 32), dim3(256), 0, stream>>>(w_k,  wt_k);
  prep_weights<<<dim3(DM / 32, DM / 32), dim3(256), 0, stream>>>(w_v,  wt_v);
  prep_weights<<<dim3(DM / 32, DM / 32), dim3(256), 0, stream>>>(w_fc, wt_fc);

  // 1) projections
  proj_gemm<0><<<dim3(MTOT / 128, DM / 64), blk, 0, stream>>>(q, wt_q, Qh);
  proj_gemm<0><<<dim3(MTOT / 128, DM / 64), blk, 0, stream>>>(k, wt_k, Kh);
  proj_gemm<1><<<dim3(MTOT / 128, DM / 64), blk, 0, stream>>>(v, wt_v, Vt);

  // 2) attention scores
  attn_scores<<<dim3(LSEQ / 128, LSEQ / 64, BATCH * NH), blk, 0, stream>>>(Qh, Kh, attn);

  // 3) softmax (in place in d_out attn region)
  softmax_rows<<<dim3(BATCH * NH * LSEQ), dim3(256), 0, stream>>>(attn);

  // 4) attn @ V
  attn_v<<<dim3(LSEQ / 128, 1, BATCH * NH), blk, 0, stream>>>(attn, Vt, O);

  // 5) FC + residual
  fc_gemm<<<dim3(MTOT / 128, DM / 64), blk, 0, stream>>>(O, wt_fc, v, Y);

  // 6) LayerNorm -> out
  layernorm_rows<<<dim3(MTOT), dim3(256), 0, stream>>>(Y, ln_g, ln_b, out);
}
